// AntisymmetricLayer_82102594830857
// MI455X (gfx1250) — matrix operand order repaired and checked by reading, not compiled, not run on hardware
//
#include <hip/hip_runtime.h>

// Antisymmetric RNN for MI455X (gfx1250, wave32, WMMA).
// B=128, T=1024, D=128, U=256.
//
//  K0 prep : M = W - W^T - gamma*I  and V, bf16, pre-swizzled into the WMMA
//            B-operand fragment layout (lane-major, 16 elems/lane).
//  K1 proj : H = inputs @ V + bias, bf16, stored in WMMA *C-fragment* layout
//            [t][btile][ntile][lane][8] so each lane stores/loads one b128.
//  K2 scan : 8 blocks (16 batch rows) x 16 waves (one 16-col N-tile each).
//            M fragments register-resident; state fp32 in accumulator layout;
//            bf16 state shadow in LDS -> A-fragments are 2x ds_load_b128/k-block.
//            h_t enters as the C operand of the first WMMA (no add chain).

#define EPS_   0.01f
#define GAMMA_ 0.01f
#define B_ 128
#define T_ 1024
#define D_ 128
#define U_ 256

typedef __attribute__((ext_vector_type(16))) __bf16 v16bf;
typedef __attribute__((ext_vector_type(8)))  __bf16 v8bf;
typedef __attribute__((ext_vector_type(8)))  float  v8f;

// workspace layout (bytes)
#define MFRAG_OFF 0                  // 16 nt * 8 kb * 32 lanes * 16 elems bf16 = 128 KB
#define VFRAG_OFF (128*1024)         // 16 nt * 4 kb * 32 lanes * 16 elems bf16 =  64 KB
#define H_OFF     (256*1024)         // [T][8][16][32][8] bf16 = 64 MB (C-fragment order)

static __device__ __forceinline__ v16bf join8(v8bf lo, v8bf hi) {
  v16bf r;
#pragma unroll
  for (int i = 0; i < 8; ++i) { r[i] = lo[i]; r[i + 8] = hi[i]; }
  return r;
}

static __device__ __forceinline__ float fast_tanh(float x) {
  x = fminf(fmaxf(x, -10.f), 10.f);
  float e = __expf(2.f * x);           // v_exp_f32 (TRANS), co-executes with WMMA
  return (e - 1.f) / (e + 1.f);
}

// ---------------------------------------------------------------------------
// K0: bf16 fragment-layout copies of M and V.
// B-operand layout (16-bit, 32x16): lane<16 -> N=lane, K=0..7,16..23; lane>=16 ->
// N=lane-16, K=8..15,24..31.  k = kb*32 + 8*(lane>>4) + (i&7) + 16*(i>>3), n = nt*16 + (lane&15).
// ---------------------------------------------------------------------------
__global__ void asym_prep_kernel(const float* __restrict__ W, const float* __restrict__ V,
                                 unsigned short* __restrict__ mfrag_u,
                                 unsigned short* __restrict__ vfrag_u) {
  __bf16* mfrag = (__bf16*)mfrag_u;
  __bf16* vfrag = (__bf16*)vfrag_u;
  int idx = blockIdx.x * blockDim.x + threadIdx.x;
  const int MF = 16 * 8 * 32 * 16;
  const int VF = 16 * 4 * 32 * 16;
  if (idx < MF) {
    int i = idx & 15, lane = (idx >> 4) & 31, kb = (idx >> 9) & 7, nt = idx >> 12;
    int k = kb * 32 + 8 * (lane >> 4) + (i & 7) + 16 * (i >> 3);
    int n = nt * 16 + (lane & 15);
    float v = W[k * U_ + n] - W[n * U_ + k] - (k == n ? GAMMA_ : 0.f);
    mfrag[idx] = (__bf16)v;
  } else if (idx < MF + VF) {
    int j = idx - MF;
    int i = j & 15, lane = (j >> 4) & 31, kb = (j >> 9) & 3, nt = j >> 11;
    int k = kb * 32 + 8 * (lane >> 4) + (i & 7) + 16 * (i >> 3);   // k = d (0..127)
    int n = nt * 16 + (lane & 15);
    vfrag[j] = (__bf16)V[k * U_ + n];
  }
}

// ---------------------------------------------------------------------------
// K1: H (C-fragment layout) = inputs @ V + bias.
// grid (B/16, T), 512 threads = 16 waves; wave w owns N-tile w.
// ---------------------------------------------------------------------------
__global__ void __launch_bounds__(512)
asym_proj_kernel(const float* __restrict__ x, const float* __restrict__ bias,
                 const unsigned short* __restrict__ vfrag_u, unsigned short* __restrict__ H_u) {
  const __bf16* vfrag = (const __bf16*)vfrag_u;
  __bf16* H = (__bf16*)H_u;
  __shared__ __align__(16) __bf16 xs[16][128];   // 4 KB staged A tile (bf16)

  int t  = blockIdx.y;
  int bt = blockIdx.x;
  int b0 = bt * 16;
  int tid = threadIdx.x;
#pragma unroll
  for (int e = 0; e < 4; ++e) {                  // 2048 elems / 512 threads
    int idx = tid + e * 512;
    int m = idx >> 7, d = idx & 127;
    xs[m][d] = (__bf16)x[((size_t)(b0 + m)) * T_ * D_ + (size_t)t * D_ + d];
  }
  __syncthreads();

  int wave = tid >> 5, lane = tid & 31;
  int hh = lane >> 4, nl = lane & 15;

  v8f acc0 = {}, acc1 = {};
#pragma unroll
  for (int kb = 0; kb < 4; ++kb) {
    // A fragment: two ds_load_b128
    const __bf16* p = &xs[nl][kb * 32 + 8 * hh];
    v16bf a = join8(*(const v8bf*)p, *(const v8bf*)(p + 16));
    // B fragment: 32B/lane contiguous from pre-swizzled V
    const __bf16* q = vfrag + (((size_t)wave * 4 + kb) * 32 + lane) * 16;
    v16bf b = join8(*(const v8bf*)q, *(const v8bf*)(q + 8));
    if (kb & 1) acc1 = __builtin_amdgcn_wmma_f32_16x16x32_bf16(false, a, false, b, (short)0, acc1, false, false);
    else        acc0 = __builtin_amdgcn_wmma_f32_16x16x32_bf16(false, a, false, b, (short)0, acc0, false, false);
  }
  float bv = bias[wave * 16 + nl];               // same n for all 8 accum rows
  v8bf hb;
#pragma unroll
  for (int r = 0; r < 8; ++r) hb[r] = (__bf16)(acc0[r] + acc1[r] + bv);
  // one b128 store per lane, C-fragment order: [t][bt][wave][lane][8]
  *(v8bf*)(H + ((((size_t)t * 8 + bt) * 16 + wave) * 32 + lane) * 8) = hb;
}

// ---------------------------------------------------------------------------
// K2: the serial scan.  8 blocks x 512 threads (16 waves).
// ---------------------------------------------------------------------------
__global__ void __launch_bounds__(512)
asym_scan_kernel(const float* __restrict__ x0, const unsigned short* __restrict__ mfrag_u,
                 const unsigned short* __restrict__ H_u, float* __restrict__ out) {
  const __bf16* mfrag = (const __bf16*)mfrag_u;
  const __bf16* H = (const __bf16*)H_u;
  __shared__ __align__(16) __bf16 sA[16][256];   // 8 KB bf16 state shadow (A operand)

  int bt = blockIdx.x;
  int b0 = bt * 16;
  int tid = threadIdx.x;
  int wave = tid >> 5, lane = tid & 31;
  int hh = lane >> 4, nl = lane & 15;
  int n0 = wave * 16;

  // register-resident M fragments: 8 x v16bf = 64 VGPRs, loaded once
  v16bf mf[8];
#pragma unroll
  for (int kb = 0; kb < 8; ++kb) {
    const __bf16* q = mfrag + (((size_t)wave * 8 + kb) * 32 + lane) * 16;
    mf[kb] = join8(*(const v8bf*)q, *(const v8bf*)(q + 8));
  }

  // fp32 master state in accumulator layout: element r -> (m=r+8*hh, n=n0+nl)
  float s0 = x0[n0 + nl];
  v8f state;
#pragma unroll
  for (int r = 0; r < 8; ++r) state[r] = s0;

  for (int idx = tid; idx < 16 * 256; idx += 512)
    sA[idx >> 8][idx & 255] = (__bf16)x0[idx & 255];
  __syncthreads();

  // per-lane H fragment pointer for this (bt, wave, lane); advances 64KB per t
  const __bf16* hp = H + ((((size_t)0 * 8 + bt) * 16 + wave) * 32 + lane) * 8;
  const size_t hstep = (size_t)8 * 16 * 32 * 8;  // elems per t across all blocks

  for (int t = 0; t < T_; ++t) {
    // h_t fragment: ONE global_load_b128 per lane; becomes WMMA C operand
    v8bf hb = *(const v8bf*)hp;
    if (t + 1 < T_) __builtin_prefetch(hp + hstep, 0, 1);   // global_prefetch_b8
    v8f hv;
#pragma unroll
    for (int r = 0; r < 8; ++r) hv[r] = (float)hb[r];

    // state @ M + h : A-fragments from LDS (2x ds_load_b128 per k-block),
    // two independent accumulator chains; chain0 seeded with h (C operand).
    v8f acc0 = hv, acc1 = {};
#pragma unroll
    for (int kb = 0; kb < 8; ++kb) {
      const __bf16* p = &sA[nl][kb * 32 + 8 * hh];
      v16bf a = join8(*(const v8bf*)p, *(const v8bf*)(p + 16));
      if (kb & 1) acc1 = __builtin_amdgcn_wmma_f32_16x16x32_bf16(false, a, false, mf[kb], (short)0, acc1, false, false);
      else        acc0 = __builtin_amdgcn_wmma_f32_16x16x32_bf16(false, a, false, mf[kb], (short)0, acc0, false, false);
    }

    __syncthreads();   // all A-fragment reads done before the shadow is overwritten

#pragma unroll
    for (int r = 0; r < 8; ++r) {
      float z  = acc0[r] + acc1[r];
      float ns = state[r] + EPS_ * fast_tanh(z);
      state[r] = ns;
      out[((size_t)(b0 + r + 8 * hh) * T_ + t) * U_ + n0 + nl] = ns;  // [B][T][U] f32
      sA[r + 8 * hh][n0 + nl] = (__bf16)ns;
    }
    __syncthreads();   // new state visible before next step's reads
    hp += hstep;
  }
}

// ---------------------------------------------------------------------------
extern "C" void kernel_launch(void* const* d_in, const int* in_sizes, int n_in,
                              void* d_out, int out_size, void* d_ws, size_t ws_size,
                              hipStream_t stream) {
  const float* inputs = (const float*)d_in[0];
  const float* V      = (const float*)d_in[1];
  const float* W      = (const float*)d_in[2];
  const float* bias   = (const float*)d_in[3];
  const float* x0     = (const float*)d_in[4];
  float* out = (float*)d_out;

  unsigned char* ws = (unsigned char*)d_ws;
  unsigned short* mfrag = (unsigned short*)(ws + MFRAG_OFF);
  unsigned short* vfrag = (unsigned short*)(ws + VFRAG_OFF);
  unsigned short* H     = (unsigned short*)(ws + H_OFF);

  const int prep_elems = 16 * 8 * 32 * 16 + 16 * 4 * 32 * 16;
  asym_prep_kernel<<<(prep_elems + 255) / 256, 256, 0, stream>>>(W, V, mfrag, vfrag);
  asym_proj_kernel<<<dim3(B_ / 16, T_), 512, 0, stream>>>(inputs, bias, vfrag, H);
  asym_scan_kernel<<<B_ / 16, 512, 0, stream>>>(x0, mfrag, H, out);
}
